// FrequencyAttention3_2250562863502
// MI455X (gfx1250) — matrix operand order repaired and checked by reading, not compiled, not run on hardware
//
#include <hip/hip_runtime.h>
#include <hip/hip_bf16.h>
#include <math.h>

// Problem constants (B=8, L=4096, D=512, K=4)
#define B_DIM  8
#define L_DIM  4096
#define LOG2L  12
#define D_DIM  512
#define NFREQ  2049          // L/2 + 1
#define LN_EPS 1e-5f
#define MAXF   32            // cap on kept product frequencies

typedef __attribute__((ext_vector_type(16))) __bf16 v16bf;
typedef __attribute__((ext_vector_type(8)))  float  v8f;

#define W_ELEMS (D_DIM * D_DIM)   // 262144 elements per converted weight array

// =====================================================================
// Kernel 0: convert W (512x512 f32, row-major, out[m,e]=sum_d A[m,d]W[e,d])
// into hi/lo bf16 arrays pre-swizzled into WMMA B-fragment order:
//   elem((nt,kt,lane,j,p)) = ((nt*16 + kt)*32 + lane)*16 + j*2 + p
//   value = W[e = nt*16 + (lane&15)][k = kt*32 + (lane>>4)*16 + 2j + p]
// so a lane's 16-element fragment is 32 contiguous bytes.
// =====================================================================
__global__ __launch_bounds__(256)
void conv_w(const float* __restrict__ W, __bf16* __restrict__ Whi,
            __bf16* __restrict__ Wlo)
{
  int p = blockIdx.x * 256 + threadIdx.x;        // pair index, 131072 total
  int j    = p & 7;
  int lane = (p >> 3) & 31;
  int kt   = (p >> 8) & 15;
  int nt   = p >> 12;
  int e = nt * 16 + (lane & 15);
  int k = kt * 32 + (lane >> 4) * 16 + 2 * j;
  float w0 = W[(size_t)e * D_DIM + k];
  float w1 = W[(size_t)e * D_DIM + k + 1];
  __bf16 h0 = (__bf16)w0, h1 = (__bf16)w1;
  Whi[2 * p]     = h0;
  Whi[2 * p + 1] = h1;
  Wlo[2 * p]     = (__bf16)(w0 - (float)h0);
  Wlo[2 * p + 1] = (__bf16)(w1 - (float)h1);
}

// =====================================================================
// Kernel 1: out = A * W^T + bias.
// One block = 16 M-rows x all 512 N-cols. A tile (16x512) loaded from HBM
// once, split hi/lo once, staged in LDS in WMMA A-fragment order.
// 8 waves x 4 N-tiles each; inner loop = 2x ds_load_b128-pair + B frags
// from L2 + 12 WMMAs per K-step (3-term split precision).
// transpose_store: 0 -> out[m,e] ([B,L,D]);  1 -> outT[(b*D+e)*L + l]
// =====================================================================
__global__ __launch_bounds__(256)
void gemm_qkv(const float* __restrict__ A,
              const __bf16* __restrict__ Whi, const __bf16* __restrict__ Wlo,
              const float* __restrict__ bias, float* __restrict__ out,
              int transpose_store)
{
  __shared__ __attribute__((aligned(32))) __bf16 sAhi[16 * D_DIM];  // 16 KB
  __shared__ __attribute__((aligned(32))) __bf16 sAlo[16 * D_DIM];  // 16 KB

  const int tid   = threadIdx.x;
  const int lane  = tid & 31;
  const int wave  = tid >> 5;
  const int half  = lane >> 4;
  const int l15   = lane & 15;
  const int tileM = blockIdx.x * 16;

  // ---- stage A tile: coalesced f32 load, hi/lo split, fragment-order LDS ----
  for (int idx = tid; idx < 16 * D_DIM; idx += 256) {
    int k = idx & (D_DIM - 1);
    int r = idx >> 9;                      // row within tile
    float a = A[(size_t)(tileM + r) * D_DIM + k];
    int ko = k & 31, kt = k >> 5;
    int hf = (ko >> 4) & 1;                // lane half that owns this K
    int j  = (ko & 15) >> 1;
    int el = kt * 512 + (hf * 16 + r) * 16 + j * 2 + (ko & 1);
    __bf16 h = (__bf16)a;
    sAhi[el] = h;
    sAlo[el] = (__bf16)(a - (float)h);
  }
  __syncthreads();

  // ---- compute: wave handles N-tiles wave, wave+8, wave+16, wave+24 ----
  v8f c[4] = {{}, {}, {}, {}};
#pragma unroll 2
  for (int kt = 0; kt < 16; ++kt) {
    v16bf ahi = *(const v16bf*)&sAhi[kt * 512 + lane * 16];
    v16bf alo = *(const v16bf*)&sAlo[kt * 512 + lane * 16];
#pragma unroll
    for (int t = 0; t < 4; ++t) {
      int nt = wave + t * 8;
      size_t base = ((size_t)(nt * 16 + kt) * 32 + lane) * 16;
      v16bf bhi = *(const v16bf*)&Whi[base];
      v16bf blo = *(const v16bf*)&Wlo[base];
      c[t] = __builtin_amdgcn_wmma_f32_16x16x32_bf16(false, ahi, false, bhi, (short)0, c[t], false, false);
      c[t] = __builtin_amdgcn_wmma_f32_16x16x32_bf16(false, ahi, false, blo, (short)0, c[t], false, false);
      c[t] = __builtin_amdgcn_wmma_f32_16x16x32_bf16(false, alo, false, bhi, (short)0, c[t], false, false);
    }
  }

  // ---- epilogue ----
#pragma unroll
  for (int t = 0; t < 4; ++t) {
    int nt   = wave + t * 8;
    int erow = nt * 16 + l15;
    float be = bias[erow];
    if (transpose_store) {
      const int b     = tileM >> LOG2L;              // 16-row tiles never cross b
      const int lbase = (tileM & (L_DIM - 1)) + half * 8;
      float* col = out + ((size_t)b * D_DIM + erow) * L_DIM + lbase;
#pragma unroll
      for (int r = 0; r < 8; ++r) col[r] = c[t][r] + be;   // 32B contiguous/lane
    } else {
#pragma unroll
      for (int r = 0; r < 8; ++r) {
        int m = tileM + half * 8 + r;
        out[(size_t)m * D_DIM + erow] = c[t][r] + be;
      }
    }
  }
}

// =====================================================================
// Kernel 2: per (b,d) column — packed complex FFT of (k + i*q), Hermitian
// split, top-4 amplitude thresholds, sparse product spectrum, direct
// <=MAXF-term inverse transform. Writes attT over the kT buffer (safe:
// this WG fully read its column before writing it).
// =====================================================================
__device__ __forceinline__ void ins4(float4& t, float v) {
  if (v > t.x)      { t.w = t.z; t.z = t.y; t.y = t.x; t.x = v; }
  else if (v > t.y) { t.w = t.z; t.z = t.y; t.y = v; }
  else if (v > t.z) { t.w = t.z; t.z = v; }
  else if (v > t.w) { t.w = v; }
}

__global__ __launch_bounds__(256)
void freq_attn(const float* __restrict__ kT, const float* __restrict__ qT,
               float* __restrict__ attT)
{
  __shared__ float2 Z[L_DIM];          // 32 KB
  __shared__ float4 topA[256];
  __shared__ float4 topB[256];
  __shared__ float  lr[MAXF], li[MAXF];
  __shared__ int    lfreq[MAXF];
  __shared__ int    nf;

  const int tid = threadIdx.x;
  const int sig = blockIdx.x;                   // b*D + d
  const float* kcol = kT + (size_t)sig * L_DIM;
  const float* qcol = qT + (size_t)sig * L_DIM;

  for (int l = tid; l < L_DIM; l += 256) {
    int r = __brev((unsigned)l) >> (32 - LOG2L);
    Z[r] = make_float2(kcol[l], qcol[l]);
  }
  if (tid == 0) nf = 0;
  __syncthreads();

  // in-place radix-2 DIT forward FFT (e^{-i2*pi*nk/N})
  for (int s = 1; s <= LOG2L; ++s) {
    const int halfm = 1 << (s - 1);
    const float ang = -3.14159265358979323846f / (float)halfm;
    for (int j = tid; j < (L_DIM >> 1); j += 256) {
      int pos = j & (halfm - 1);
      int i0  = ((j >> (s - 1)) << s) + pos;
      int i1  = i0 + halfm;
      float sn, cs;
      __sincosf(ang * (float)pos, &sn, &cs);
      float2 u = Z[i0], v = Z[i1];
      float tr = cs * v.x - sn * v.y;
      float ti = cs * v.y + sn * v.x;
      Z[i0] = make_float2(u.x + tr, u.y + ti);
      Z[i1] = make_float2(u.x - tr, u.y - ti);
    }
    __syncthreads();
  }

  // top-4 of |Kf|^2 and |Qf|^2 (squared magnitude: same mask)
  float4 t4k = make_float4(-1.f, -1.f, -1.f, -1.f);
  float4 t4q = make_float4(-1.f, -1.f, -1.f, -1.f);
  for (int f = tid; f < NFREQ; f += 256) {
    float2 zf = Z[f];
    float2 zc = Z[(L_DIM - f) & (L_DIM - 1)];
    float kr = 0.5f * (zf.x + zc.x), ki = 0.5f * (zf.y - zc.y);
    float qr = 0.5f * (zf.y + zc.y), qi = 0.5f * (zc.x - zf.x);
    ins4(t4k, kr * kr + ki * ki);
    ins4(t4q, qr * qr + qi * qi);
  }
  topA[tid] = t4k;
  topB[tid] = t4q;
  __syncthreads();
  for (int s = 128; s >= 1; s >>= 1) {
    if (tid < s) {
      float4 a = topA[tid + s], t = topA[tid];
      ins4(t, a.x); ins4(t, a.y); ins4(t, a.z); ins4(t, a.w);
      topA[tid] = t;
      float4 b = topB[tid + s], u = topB[tid];
      ins4(u, b.x); ins4(u, b.y); ins4(u, b.z); ins4(u, b.w);
      topB[tid] = u;
    }
    __syncthreads();
  }
  const float thrK = topA[0].w;
  const float thrQ = topB[0].w;

  // sparse product spectrum: P = Qf * conj(Kf) where both masks keep f
  for (int f = tid; f < NFREQ; f += 256) {
    float2 zf = Z[f];
    float2 zc = Z[(L_DIM - f) & (L_DIM - 1)];
    float kr = 0.5f * (zf.x + zc.x), ki = 0.5f * (zf.y - zc.y);
    float qr = 0.5f * (zf.y + zc.y), qi = 0.5f * (zc.x - zf.x);
    float ak = kr * kr + ki * ki, aq = qr * qr + qi * qi;
    if (ak >= thrK && aq >= thrQ) {
      float pr = qr * kr + qi * ki;
      float pi = qi * kr - qr * ki;
      float sc = ((f == 0) || (f == (L_DIM / 2))) ? 1.0f : 2.0f;
      sc *= (1.0f / (float)L_DIM);
      int idx = atomicAdd(&nf, 1);
      if (idx < MAXF) { lfreq[idx] = f; lr[idx] = sc * pr; li[idx] = sc * pi; }
    }
  }
  __syncthreads();

  const int n = (nf < MAXF) ? nf : MAXF;
  float* ocol = attT + (size_t)sig * L_DIM;
  const float w0 = 6.283185307179586477f / (float)L_DIM;
  for (int l = tid; l < L_DIM; l += 256) {
    float acc = 0.f;
    for (int j = 0; j < n; ++j) {
      int m = (lfreq[j] * l) & (L_DIM - 1);   // exact phase reduction
      float sn, cs;
      __sincosf(w0 * (float)m, &sn, &cs);
      acc += lr[j] * cs - li[j] * sn;
    }
    ocol[l] = acc;
  }
}

// =====================================================================
// Kernel 3: per (b,l) row — softmax over D, context = att*v, residual,
// LayerNorm. LDS tile un-transposes attT; one wave32 per two rows.
// =====================================================================
__global__ __launch_bounds__(256)
void softmax_ln(const float* __restrict__ attT, const float* __restrict__ vbuf,
                const float* __restrict__ query, const float* __restrict__ ln_w,
                const float* __restrict__ ln_b, float* __restrict__ out)
{
  __shared__ float att_s[16][D_DIM + 4];   // 33 KB
  const int tid  = threadIdx.x;
  const int lane = tid & 31;
  const int wave = tid >> 5;
  const int b  = blockIdx.x >> 8;
  const int l0 = (blockIdx.x & 255) * 16;

  for (int idx = tid; idx < 16 * D_DIM; idx += 256) {
    int dd = idx >> 4, ll = idx & 15;
    att_s[ll][dd] = attT[((size_t)b * D_DIM + dd) * L_DIM + l0 + ll];
  }
  __syncthreads();

#pragma unroll
  for (int rr = 0; rr < 2; ++rr) {
    const int l  = wave * 2 + rr;
    const int gl = l0 + l;
    float vals[16];
    float mx = -1e30f;
#pragma unroll
    for (int i = 0; i < 16; ++i) {
      vals[i] = att_s[l][lane + 32 * i];
      mx = fmaxf(mx, vals[i]);
    }
    for (int o = 16; o >= 1; o >>= 1) mx = fmaxf(mx, __shfl_xor(mx, o, 32));
    float sum = 0.f;
#pragma unroll
    for (int i = 0; i < 16; ++i) { vals[i] = __expf(vals[i] - mx); sum += vals[i]; }
    for (int o = 16; o >= 1; o >>= 1) sum += __shfl_xor(sum, o, 32);
    const float inv = 1.0f / sum;

    const size_t base = ((size_t)b * L_DIM + gl) * D_DIM;
    float x[16], s1 = 0.f, s2 = 0.f;
#pragma unroll
    for (int i = 0; i < 16; ++i) {
      int dd = lane + 32 * i;
      float xv = query[base + dd] + vals[i] * inv * vbuf[base + dd];
      x[i] = xv; s1 += xv; s2 += xv * xv;
    }
    for (int o = 16; o >= 1; o >>= 1) {
      s1 += __shfl_xor(s1, o, 32);
      s2 += __shfl_xor(s2, o, 32);
    }
    const float mu   = s1 * (1.0f / D_DIM);
    const float var  = s2 * (1.0f / D_DIM) - mu * mu;
    const float rstd = rsqrtf(var + LN_EPS);
#pragma unroll
    for (int i = 0; i < 16; ++i) {
      int dd = lane + 32 * i;
      out[base + dd] = (x[i] - mu) * rstd * ln_w[dd] + ln_b[dd];
    }
  }
}

// =====================================================================
// Host launcher.  Workspace layout (needs 193 MB):
//   kT   [B,D,L] f32 @ 0        (reused as attT after freq_attn reads it)
//   qT   [B,D,L] f32 @ 64 MB
//   v    [B,L,D] f32 @ 128 MB
//   Whi/Wlo bf16 swizzled       @ 192 MB (1 MB, reused per projection)
// =====================================================================
extern "C" void kernel_launch(void* const* d_in, const int* in_sizes, int n_in,
                              void* d_out, int out_size, void* d_ws, size_t ws_size,
                              hipStream_t stream)
{
  (void)in_sizes; (void)n_in; (void)out_size; (void)ws_size;
  const float* key   = (const float*)d_in[0];
  const float* value = (const float*)d_in[1];
  const float* query = (const float*)d_in[2];
  const float* Wk    = (const float*)d_in[3];
  const float* bk    = (const float*)d_in[4];
  const float* Wv    = (const float*)d_in[5];
  const float* bv    = (const float*)d_in[6];
  const float* Wq    = (const float*)d_in[7];
  const float* bq    = (const float*)d_in[8];
  const float* ln_w  = (const float*)d_in[9];
  const float* ln_b  = (const float*)d_in[10];
  float* out = (float*)d_out;

  const size_t SZ = (size_t)B_DIM * L_DIM * D_DIM;
  float*  kT   = (float*)d_ws;
  float*  qT   = kT + SZ;
  float*  vbuf = qT + SZ;
  __bf16* whi  = (__bf16*)(vbuf + SZ);
  __bf16* wlo  = whi + W_ELEMS;

  const int M_BLKS = (B_DIM * L_DIM) / 16;          // 2048
  const int CW_BLKS = (W_ELEMS / 2) / 256;          // 512

  conv_w<<<CW_BLKS, 256, 0, stream>>>(Wk, whi, wlo);
  gemm_qkv<<<M_BLKS, 256, 0, stream>>>(key, whi, wlo, bk, kT, 1);

  conv_w<<<CW_BLKS, 256, 0, stream>>>(Wq, whi, wlo);
  gemm_qkv<<<M_BLKS, 256, 0, stream>>>(query, whi, wlo, bq, qT, 1);

  conv_w<<<CW_BLKS, 256, 0, stream>>>(Wv, whi, wlo);
  gemm_qkv<<<M_BLKS, 256, 0, stream>>>(value, whi, wlo, bv, vbuf, 0);

  freq_attn<<<B_DIM * D_DIM, 256, 0, stream>>>(kT, qT, /*attT=*/kT);

  softmax_ln<<<B_DIM * (L_DIM / 16), 256, 0, stream>>>(kT, vbuf, query, ln_w, ln_b, out);
}
